// GNN_10883447128068
// MI455X (gfx1250) — matrix operand order repaired and checked by reading, not compiled, not run on hardware
//
#include <hip/hip_runtime.h>

// Problem constants (match reference setup_inputs()).
#define NN 50000      // nodes
#define EE 640000     // edges
#define GG 128        // graphs
#define HH 128        // hidden channels
#define F_NODE 64
#define F_EDGE 16
#define F_LIG 200
#define F_POC 100
#define K_LIG_PAD 224 // 200 -> next multiple of 32
#define K_POC_PAD 128 // 100 -> next multiple of 32
#define NSTACK 3

typedef __attribute__((ext_vector_type(16))) _Float16 v16h;
typedef __attribute__((ext_vector_type(4)))  _Float16 v4h;
typedef __attribute__((ext_vector_type(8)))  float    v8f;

// ---------------------------------------------------------------------------
// Y[M x 128] (row stride ldY) = X[M x K] @ W[K x 128] + bias
// K: compile-time multiple of 32; M % 16 == 0; Hout fixed at 128.
// One wave32 computes a full 16x128 band: the A fragment (16x32 f16) is
// loaded ONCE per K-step and fed to 8 independent v_wmma_f32_16x16x32_f16
// (one per 16-column tile, separate accumulators -> no D->A/B hazards).
// A/B fragments use contiguous k order (k = 16g + s). OUT_H stores D as f16 to halve
// downstream gather traffic.
// ---------------------------------------------------------------------------
template<int K, bool OUT_H>
__global__ void gnn_gemm_wmma(const float* __restrict__ X,
                              const float* __restrict__ W,
                              const float* __restrict__ bias,
                              void* __restrict__ Yv,
                              int M, int ldY)
{
  const int lane = threadIdx.x & 31;
  const int wave = threadIdx.x >> 5;
  const int band = blockIdx.x * (blockDim.x >> 5) + wave;
  if (band >= (M >> 4)) return;          // wave-uniform: EXEC stays all-1s

  const int g  = lane >> 4;              // half-wave group (0/1)
  const int lm = lane & 15;
  const int row = band * 16 + lm;        // A-matrix row held by this lane

  const float* __restrict__ Xr = X + (size_t)row * K;

  v8f c[8];
#pragma unroll
  for (int t = 0; t < 8; ++t)
#pragma unroll
    for (int r = 0; r < 8; ++r) c[t][r] = 0.0f;

  for (int k0 = 0; k0 < K; k0 += 32) {
    // A 16x32 f16 fragment: half h -> K = k0+16g+h
    const int ka = k0 + 16 * g;
    const float4 a0 = *(const float4*)(Xr + ka);
    const float4 a1 = *(const float4*)(Xr + ka + 4);
    const float4 a2 = *(const float4*)(Xr + ka + 8);
    const float4 a3 = *(const float4*)(Xr + ka + 12);
    v16h a;
    a[0]  = (_Float16)a0.x; a[1]  = (_Float16)a0.y;
    a[2]  = (_Float16)a0.z; a[3]  = (_Float16)a0.w;
    a[4]  = (_Float16)a1.x; a[5]  = (_Float16)a1.y;
    a[6]  = (_Float16)a1.z; a[7]  = (_Float16)a1.w;
    a[8]  = (_Float16)a2.x; a[9]  = (_Float16)a2.y;
    a[10] = (_Float16)a2.z; a[11] = (_Float16)a2.w;
    a[12] = (_Float16)a3.x; a[13] = (_Float16)a3.y;
    a[14] = (_Float16)a3.z; a[15] = (_Float16)a3.w;

    // For each 16-column tile: B 32x16 f16 fragment (half h -> K = k0+16g+h)
    const size_t kbase = (size_t)(k0 + 16 * g) * HH + lm;
#pragma unroll
    for (int t = 0; t < 8; ++t) {
      const float* __restrict__ Wc = W + kbase + t * 16;
      v16h b;
#pragma unroll
      for (int hh = 0; hh < 16; ++hh)
        b[hh] = (_Float16)Wc[hh * HH];
      c[t] = __builtin_amdgcn_wmma_f32_16x16x32_f16(false, a, false, b,
                                                    (short)0, c[t], false, false);
    }
  }

#pragma unroll
  for (int t = 0; t < 8; ++t) {
    const int col = t * 16 + lm;         // B/C/D column held by this lane
    const float bv = bias[col];
#pragma unroll
    for (int r = 0; r < 8; ++r) {        // C/D: VGPR r -> row r + 8g
      const int orow = band * 16 + r + 8 * g;
      if constexpr (OUT_H)
        ((_Float16*)Yv)[(size_t)orow * ldY + col] = (_Float16)(c[t][r] + bv);
      else
        ((float*)Yv)[(size_t)orow * ldY + col] = c[t][r] + bv;
    }
  }
}

// ---------------------------------------------------------------------------
// Fused edge-message + scatter-add:
//   agg[dst] += hW[src] + edge_attr[e] @ W_edge + b_edge
// W_edge (16x128 = 8KB) and b_edge cached in LDS. One thread handles one
// (edge, 4-channel) quad: v4h (8B) gather of f16 hW, 64 LDS-fed FMAs,
// 4 global f32 atomics.
// ---------------------------------------------------------------------------
__global__ void gnn_edge_scatter(const _Float16* __restrict__ hW,
                                 const int* __restrict__ edge_index,
                                 const float* __restrict__ edge_attr,
                                 const float* __restrict__ W_edge,
                                 const float* __restrict__ b_edge,
                                 float* __restrict__ agg)
{
  __shared__ float sW[F_EDGE * HH];
  __shared__ float sB[HH];
  for (int i = threadIdx.x; i < F_EDGE * HH; i += blockDim.x) sW[i] = W_edge[i];
  if (threadIdx.x < HH) sB[threadIdx.x] = b_edge[threadIdx.x];
  __syncthreads();

  const int idx = blockIdx.x * blockDim.x + threadIdx.x;
  if (idx >= EE * 32) return;
  const int e   = idx >> 5;
  const int c0  = (idx & 31) << 2;
  const int src = edge_index[e];
  const int dst = edge_index[EE + e];

  float ea[F_EDGE];
  const float4* eap = (const float4*)(edge_attr + (size_t)e * F_EDGE);
#pragma unroll
  for (int q = 0; q < 4; ++q) {
    const float4 v = eap[q];
    ea[4 * q + 0] = v.x; ea[4 * q + 1] = v.y;
    ea[4 * q + 2] = v.z; ea[4 * q + 3] = v.w;
  }

  const v4h hv = *(const v4h*)(hW + (size_t)src * HH + c0);
  float v0 = (float)hv[0] + sB[c0 + 0];
  float v1 = (float)hv[1] + sB[c0 + 1];
  float v2 = (float)hv[2] + sB[c0 + 2];
  float v3 = (float)hv[3] + sB[c0 + 3];
#pragma unroll
  for (int k = 0; k < F_EDGE; ++k) {
    const float* wr = &sW[k * HH + c0];
    v0 = fmaf(ea[k], wr[0], v0);
    v1 = fmaf(ea[k], wr[1], v1);
    v2 = fmaf(ea[k], wr[2], v2);
    v3 = fmaf(ea[k], wr[3], v3);
  }
  float* ap = agg + (size_t)dst * HH + c0;
  atomicAdd(ap + 0, v0);
  atomicAdd(ap + 1, v1);
  atomicAdd(ap + 2, v2);
  atomicAdd(ap + 3, v3);
}

// h = relu(agg); agg = relu(agg)  (re-seeds agg for the next identity-skip layer)
__global__ void gnn_relu_dual(float* __restrict__ agg, float* __restrict__ h, int n)
{
  const int i = blockIdx.x * blockDim.x + threadIdx.x;
  if (i < n) {
    const float v = fmaxf(agg[i], 0.0f);
    h[i]   = v;
    agg[i] = v;
  }
}

__global__ void gnn_zero(float* __restrict__ p, int n)
{
  const int i = blockIdx.x * blockDim.x + threadIdx.x;
  if (i < n) p[i] = 0.0f;
}

// Contiguous copy with zero tail: pads W [K,128] -> [Kpad,128] (row-major).
__global__ void gnn_pad_rows(const float* __restrict__ src, float* __restrict__ dst,
                             int nsrc, int ndst)
{
  const int i = blockIdx.x * blockDim.x + threadIdx.x;
  if (i < ndst) dst[i] = (i < nsrc) ? src[i] : 0.0f;
}

// Pads X [M,K] -> [M,Kpad] (zero-fill new columns).
__global__ void gnn_pad_cols(const float* __restrict__ src, float* __restrict__ dst,
                             int M, int K, int Kp)
{
  const int i = blockIdx.x * blockDim.x + threadIdx.x;
  if (i < M * Kp) {
    const int r = i / Kp;
    const int k = i - r * Kp;
    dst[i] = (k < K) ? src[(size_t)r * K + k] : 0.0f;
  }
}

// Segment-sum over graphs (and node counts via channel-0 lanes).
__global__ void gnn_pool_acc(const float* __restrict__ h, const int* __restrict__ batch,
                             float* __restrict__ sums, float* __restrict__ counts)
{
  const int i = blockIdx.x * blockDim.x + threadIdx.x;
  if (i < NN * HH) {
    const int n = i >> 7;          // HH == 128
    const int c = i & (HH - 1);
    const int b = batch[n];
    atomicAdd(&sums[(size_t)b * HH + c], h[i]);
    if (c == 0) atomicAdd(&counts[b], 1.0f);
  }
}

// pooled -> columns [0,128) of the concat buffer [G, 3H]
__global__ void gnn_pool_div(const float* __restrict__ sums,
                             const float* __restrict__ counts,
                             float* __restrict__ cat)
{
  const int i = blockIdx.x * blockDim.x + threadIdx.x;
  if (i < GG * HH) {
    const int gI = i >> 7;
    const int c  = i & (HH - 1);
    cat[(size_t)gI * (3 * HH) + c] = sums[i] / fmaxf(counts[gI], 1.0f);
  }
}

// out[g] = fused[g,:] . Wo + bo
__global__ void gnn_head(const float* __restrict__ fused, const float* __restrict__ Wo,
                         const float* __restrict__ bo, float* __restrict__ out)
{
  const int gI = blockIdx.x * blockDim.x + threadIdx.x;
  if (gI < GG) {
    float s = 0.0f;
    for (int c = 0; c < HH; ++c) s = fmaf(fused[(size_t)gI * HH + c], Wo[c], s);
    out[gI] = s + bo[0];
  }
}

__global__ void gnn_copy(const float* __restrict__ src, float* __restrict__ dst, int n)
{
  const int i = blockIdx.x * blockDim.x + threadIdx.x;
  if (i < n) dst[i] = src[i];
}

// ---------------------------------------------------------------------------
extern "C" void kernel_launch(void* const* d_in, const int* in_sizes, int n_in,
                              void* d_out, int out_size, void* d_ws, size_t ws_size,
                              hipStream_t stream)
{
  (void)in_sizes; (void)n_in; (void)ws_size;

  const float* x       = (const float*)d_in[0];
  const int*   eidx    = (const int*)  d_in[1];
  const float* eattr   = (const float*)d_in[2];
  const int*   batch   = (const int*)  d_in[3];
  const float* ligf    = (const float*)d_in[4];
  const float* pocf    = (const float*)d_in[5];
  const float* W1_msg  = (const float*)d_in[6];
  const float* b1_msg  = (const float*)d_in[7];
  const float* W1_edge = (const float*)d_in[8];
  const float* b1_edge = (const float*)d_in[9];
  const float* W1_self = (const float*)d_in[10];
  const float* b1_self = (const float*)d_in[11];
  const float* Wk_msg  = (const float*)d_in[12];
  const float* bk_msg  = (const float*)d_in[13];
  const float* Wk_edge = (const float*)d_in[14];
  const float* bk_edge = (const float*)d_in[15];
  const float* Wl      = (const float*)d_in[16];
  const float* bl      = (const float*)d_in[17];
  const float* Wp      = (const float*)d_in[18];
  const float* bp      = (const float*)d_in[19];
  const float* Wf      = (const float*)d_in[20];
  const float* bf      = (const float*)d_in[21];
  const float* Wo      = (const float*)d_in[22];
  const float* bo      = (const float*)d_in[23];

  // Workspace carve-up (float units): ~64 MB total.
  float* ws    = (float*)d_ws;
  float*    h     = ws;                                 // N*H f32
  float*    agg   = h    + (size_t)NN * HH;             // N*H f32
  _Float16* hWh   = (_Float16*)(agg + (size_t)NN * HH); // N*H f16
  float*    sums  = (float*)(hWh + (size_t)NN * HH);    // G*H
  float*    cnts  = sums + (size_t)GG * HH;             // G (contiguous after sums)
  float*    cat   = cnts + GG;                          // G*3H
  float*    fused = cat  + (size_t)GG * 3 * HH;         // G*H
  float*    ligp  = fused + (size_t)GG * HH;            // G*224
  float*    Wlp   = ligp + (size_t)GG * K_LIG_PAD;      // 224*H
  float*    pocp  = Wlp  + (size_t)K_LIG_PAD * HH;      // G*128
  float*    Wpp   = pocp + (size_t)GG * K_POC_PAD;      // 128*H

  const int T = 256;                                    // 8 waves / block
  const int gbN = ((NN >> 4) + 7) / 8;                  // node GEMM blocks (1 wave/band)
  const int gbG = ((GG >> 4) + 7) / 8;                  // graph GEMM blocks
  const int nEl = NN * HH;

  // ---- conv1 (Linear self-connection seeds agg) ----
  gnn_gemm_wmma<F_NODE, false><<<gbN, T, 0, stream>>>(x, W1_self, b1_self, agg, NN, HH);
  gnn_gemm_wmma<F_NODE, true ><<<gbN, T, 0, stream>>>(x, W1_msg,  b1_msg,  hWh, NN, HH);
  gnn_edge_scatter<<<(EE * 32 + T - 1) / T, T, 0, stream>>>(hWh, eidx, eattr,
                                                            W1_edge, b1_edge, agg);
  gnn_relu_dual<<<(nEl + T - 1) / T, T, 0, stream>>>(agg, h, nEl);

  // ---- stacked convs (identity self-connection: agg pre-seeded with h) ----
  for (int i = 0; i < NSTACK; ++i) {
    gnn_gemm_wmma<HH, true><<<gbN, T, 0, stream>>>(h, Wk_msg + (size_t)i * HH * HH,
                                                   bk_msg + i * HH, hWh, NN, HH);
    gnn_edge_scatter<<<(EE * 32 + T - 1) / T, T, 0, stream>>>(
        hWh, eidx, eattr, Wk_edge + (size_t)i * F_EDGE * HH, bk_edge + i * HH, agg);
    gnn_relu_dual<<<(nEl + T - 1) / T, T, 0, stream>>>(agg, h, nEl);
  }

  // ---- global mean pool ----
  gnn_zero<<<(GG * HH + GG + T - 1) / T, T, 0, stream>>>(sums, GG * HH + GG);
  gnn_pool_acc<<<(nEl + T - 1) / T, T, 0, stream>>>(h, batch, sums, cnts);
  gnn_pool_div<<<(GG * HH + T - 1) / T, T, 0, stream>>>(sums, cnts, cat);

  // ---- pad ragged-K operands to multiples of 32 ----
  gnn_pad_cols<<<(GG * K_LIG_PAD + T - 1) / T, T, 0, stream>>>(ligf, ligp, GG, F_LIG, K_LIG_PAD);
  gnn_pad_rows<<<(K_LIG_PAD * HH + T - 1) / T, T, 0, stream>>>(Wl, Wlp, F_LIG * HH, K_LIG_PAD * HH);
  gnn_pad_cols<<<(GG * K_POC_PAD + T - 1) / T, T, 0, stream>>>(pocf, pocp, GG, F_POC, K_POC_PAD);
  gnn_pad_rows<<<(K_POC_PAD * HH + T - 1) / T, T, 0, stream>>>(Wp, Wpp, F_POC * HH, K_POC_PAD * HH);

  // ---- ligand / pocket projections straight into concat column slices ----
  gnn_gemm_wmma<K_LIG_PAD, false><<<gbG, T, 0, stream>>>(ligp, Wlp, bl, cat + HH,     GG, 3 * HH);
  gnn_gemm_wmma<K_POC_PAD, false><<<gbG, T, 0, stream>>>(pocp, Wpp, bp, cat + 2 * HH, GG, 3 * HH);

  // ---- fusion (cat is [G, 384] row-major, K = 384) ----
  gnn_gemm_wmma<3 * HH, false><<<gbG, T, 0, stream>>>(cat, Wf, bf, fused, GG, HH);

  // ---- output head (out_size tells us return_embeddings deterministically) ----
  if (out_size == GG * HH) {
    gnn_copy<<<(GG * HH + T - 1) / T, T, 0, stream>>>(fused, (float*)d_out, GG * HH);
  } else {
    gnn_head<<<1, 128, 0, stream>>>(fused, Wo, bo, (float*)d_out);
  }
}
